// samba_hemo_to_ele_v1_15848429322473
// MI455X (gfx1250) — matrix operand order repaired and checked by reading, not compiled, not run on hardware
//
#include <hip/hip_runtime.h>
#include <hip/hip_bf16.h>

typedef _Float16 half_t;
typedef __attribute__((ext_vector_type(16))) _Float16 v16h;
typedef __attribute__((ext_vector_type(8)))  float    v8f;

#define WMMA_F16(a, b, c) \
  __builtin_amdgcn_wmma_f32_16x16x32_f16(false, (a), false, (b), (short)0, (c), false, false)

// ---------------- WMMA fragment loaders (contiguous k = 16*(lane/16) + i, wave32) ----------------
// A: 16x32 f16 (MxK), row-major source with leading dim ld; one 32B run/lane.
__device__ __forceinline__ v16h frag_a_ld(const half_t* base, int ld, int k0, int lane) {
  v16h a;
  const int m  = lane & 15;
  const int kb = (lane < 16) ? 0 : 16;
  const half_t* p = base + (long)m * ld + k0 + kb;
#pragma unroll
  for (int i = 0; i < 16; ++i) a[i] = p[i];
  return a;
}
// B: 32x16 f16 (KxN), source packed column-major-in-K: base[n*ld + k]; one 32B run/lane.
__device__ __forceinline__ v16h frag_b_ld(const half_t* base, int ld, int k0, int lane) {
  v16h b;
  const int n  = lane & 15;
  const int kb = (lane < 16) ? 0 : 16;
  const half_t* p = base + (long)n * ld + k0 + kb;
#pragma unroll
  for (int i = 0; i < 16; ++i) b[i] = p[i];
  return b;
}

__device__ __forceinline__ float sigmf_(float x) { return 1.0f / (1.0f + __expf(-x)); }

// ---------------- utility / packing kernels ----------------
__global__ void k_zero_h(half_t* p, long n) {
  long i = (long)blockIdx.x * blockDim.x + threadIdx.x;
  if (i < n) p[i] = (half_t)0.0f;
}

// xT[sp][b][256]: time-major, channel-contiguous, zero-padded (sp = s + 2)
__global__ void k_pack_xT(const float* __restrict__ x, half_t* __restrict__ xT) {
  long i = (long)blockIdx.x * blockDim.x + threadIdx.x;
  const long N = 1008L * 16 * 256;
  if (i >= N) return;
  int ci = (int)(i & 255);
  long r = i >> 8;
  int b = (int)(r & 15);
  int sp = (int)(r >> 4);
  int s = sp - 2;
  float v = (ci < 200 && s >= 0 && s < 1000) ? x[((long)b * 200 + ci) * 1000 + s] : 0.0f;
  xT[i] = (half_t)v;
}

// wcf[((p*5+j)*208+co)*256 + ci] = conv_w[ci][co][p+6j] (f16), zero-padded
__global__ void k_pack_convw(const float* __restrict__ w, half_t* __restrict__ wc) {
  long i = (long)blockIdx.x * blockDim.x + threadIdx.x;
  const long N = 6L * 5 * 208 * 256;
  if (i >= N) return;
  int ci = (int)(i & 255);
  long r = i >> 8;
  int co = (int)(r % 208);
  int pj = (int)(r / 208);
  int j = pj % 5, p = pj / 5;
  float v = (ci < 200 && co < 200) ? w[((long)ci * 200 + co) * 30 + (p + 6 * j)] : 0.0f;
  wc[i] = (half_t)v;
}

// generic [nr][nc] f32 -> [NR][NC] f16 zero-padded (B-fragment layout: row n holds K)
__global__ void k_pack_pad(const float* __restrict__ src, half_t* __restrict__ dst,
                           int nr, int nc, int NR, int NC) {
  long i = (long)blockIdx.x * blockDim.x + threadIdx.x;
  if (i >= (long)NR * NC) return;
  int k = (int)(i % NC);
  int n = (int)(i / NC);
  dst[i] = (n < nr && k < nc) ? (half_t)src[(long)n * nc + k] : (half_t)0.0f;
}

// ---------------- stage 1: ConvTranspose1d as 6 phase-GEMMs + GELU ----------------
// grid (13 n-tiles, 1000 s0, 6 phases), block 32 (1 wave)
__global__ void k_convt(const half_t* __restrict__ xT, const half_t* __restrict__ wcf,
                        const float* __restrict__ conv_b, float* __restrict__ y) {
  const int lane = threadIdx.x;
  const int ntile = blockIdx.x;
  const int s0 = blockIdx.y;
  const int ph = blockIdx.z;
  v8f acc = {};
#pragma unroll 1
  for (int k0 = 0; k0 < 1280; k0 += 32) {  // K = 5 taps x 256 (ci padded)
    const int j = k0 >> 8;                 // tap index
    const int sp = s0 + 4 - j;             // x time index (padded by +2)
    // A rows (batch) contiguous in ci: xT[sp][b][256]
    v16h a = frag_a_ld(xT + ((long)sp * 16) * 256 + (k0 & 255), 256, 0, lane);
    v16h b = frag_b_ld(wcf + (((long)(ph * 5 + j)) * 208 + ntile * 16) * 256 + (k0 & 255),
                       256, 0, lane);
    acc = WMMA_F16(a, b, acc);
  }
  const int co = ntile * 16 + (lane & 15);
  if (co < 200) {
    const float bias = conv_b[co];
    const int t = 6 * s0 + ph;
#pragma unroll
    for (int i = 0; i < 8; ++i) {
      int bb = i + ((lane < 16) ? 0 : 8);
      float v = acc[i] + bias;
      v = 0.5f * v * (1.0f + erff(v * 0.70710678118654752f));  // exact GELU
      y[((long)bb * 200 + co) * 6000 + t] = v;
    }
  }
}

// ---------------- stage 2: BatchNorm statistics (per channel, biased var) ----------
__global__ void k_bn_stats(const float* __restrict__ y, float* __restrict__ mu,
                           float* __restrict__ rstd) {
  __shared__ float rs[256], rs2[256];
  const int c = blockIdx.x;
  float s = 0.0f, s2 = 0.0f;
  for (int i = threadIdx.x; i < 16 * 6000; i += 256) {
    int b = i / 6000, t = i - b * 6000;
    float v = y[((long)b * 200 + c) * 6000 + t];
    s += v;
    s2 += v * v;
  }
  rs[threadIdx.x] = s;
  rs2[threadIdx.x] = s2;
  __syncthreads();
  for (int off = 128; off > 0; off >>= 1) {
    if (threadIdx.x < off) {
      rs[threadIdx.x] += rs[threadIdx.x + off];
      rs2[threadIdx.x] += rs2[threadIdx.x + off];
    }
    __syncthreads();
  }
  if (threadIdx.x == 0) {
    float m = rs[0] * (1.0f / 96000.0f);
    float var = rs2[0] * (1.0f / 96000.0f) - m * m;
    mu[c] = m;
    rstd[c] = rsqrtf(var + 1e-5f);
  }
}

// ---------------- stage 3: normalize + build f16 LSTM input sequence --------------
// inp16[t][b][224]; t=0 seeded from x_meg[:, :, 0]; t>0 from x_meg_hat[:, :, t-1]
__global__ void k_bn_apply(const float* __restrict__ y, const float* __restrict__ mu,
                           const float* __restrict__ rstd, const float* __restrict__ g,
                           const float* __restrict__ be, const float* __restrict__ xmeg,
                           half_t* __restrict__ inp16) {
  long i = (long)blockIdx.x * blockDim.x + threadIdx.x;
  const long N = 16L * 200 * 6000;
  if (i >= N) return;
  int t = (int)(i % 6000);
  long bf = i / 6000;
  int f = (int)(bf % 200);
  int b = (int)(bf / 200);
  float v = (y[i] - mu[f]) * rstd[f] * g[f] + be[f];
  if (t < 5999) inp16[((long)(t + 1) * 16 + b) * 224 + f] = (half_t)v;
  if (t == 0) inp16[(long)b * 224 + f] = (half_t)xmeg[((long)b * 200 + f) * 6000];
}

// ---------------- stage 4: precompute layer-0 input projection (big GEMM) ---------
// zpre[t*16+b][800] = inp16 @ w_ih0^T + (b_ih0 + b_hh0); grid (50, 6000), block 32
__global__ void k_zpre(const half_t* __restrict__ inp16, const half_t* __restrict__ wih0f,
                       const float* __restrict__ b_ih0, const float* __restrict__ b_hh0,
                       half_t* __restrict__ zpre) {
  const int lane = threadIdx.x;
  const int ntile = blockIdx.x;
  const int mtile = blockIdx.y;
  v8f acc = {};
  const half_t* A = inp16 + (long)mtile * 16 * 224;
  const half_t* Bp = wih0f + (long)ntile * 16 * 224;
#pragma unroll
  for (int k0 = 0; k0 < 224; k0 += 32) {
    v16h a = frag_a_ld(A, 224, k0, lane);
    v16h b = frag_b_ld(Bp, 224, k0, lane);
    acc = WMMA_F16(a, b, acc);
  }
  const int col = ntile * 16 + (lane & 15);
  const float bias = b_ih0[col] + b_hh0[col];
  half_t* zp = zpre + (long)mtile * 16 * 800;
#pragma unroll
  for (int i = 0; i < 8; ++i) {
    int m = i + ((lane < 16) ? 0 : 8);
    zp[m * 800 + col] = (half_t)(acc[i] + bias);
  }
}

// ---------------- stage 5: persistent sequential 2-layer LSTM ---------------------
// single workgroup, 16 waves; state in LDS; recurrent GEMMs via WMMA
__global__ __launch_bounds__(512) void k_lstm(
    const half_t* __restrict__ zpre, const half_t* __restrict__ whh0f,
    const half_t* __restrict__ wih1f, const half_t* __restrict__ whh1f,
    const float* __restrict__ b_ih1, const float* __restrict__ b_hh1,
    const float* __restrict__ h0, const float* __restrict__ c0,
    half_t* __restrict__ h2seq) {
  __shared__ half_t h1s[16 * 224];
  __shared__ half_t h2s[16 * 224];
  __shared__ float c1s[16 * 200];
  __shared__ float c2s[16 * 200];
  __shared__ float zbuf[16 * 800];

  const int tid = threadIdx.x;
  const int wv = tid >> 5;
  const int lane = tid & 31;

  for (int idx = tid; idx < 16 * 224; idx += 512) {
    int q = idx % 224;
    h1s[idx] = (q < 200) ? (half_t)h0[q] : (half_t)0.0f;
    h2s[idx] = (q < 200) ? (half_t)h0[200 + q] : (half_t)0.0f;
  }
  for (int idx = tid; idx < 16 * 200; idx += 512) {
    int q = idx % 200;
    c1s[idx] = c0[q];
    c2s[idx] = c0[200 + q];
  }
  __syncthreads();

  for (int t = 0; t < 6000; ++t) {
    if (t + 1 < 6000)  // warm next step's precomputed gates into cache
      __builtin_prefetch(zpre + (long)(t + 1) * 12800 + wv * 800, 0, 1);

    // ---- layer 0: zbuf = zpre[t] + h1 @ w_hh0^T ----
    const half_t* zp = zpre + (long)t * 12800;
    for (int nt = wv; nt < 50; nt += 16) {
      v8f acc = {};
      const half_t* Bp = whh0f + (long)nt * 16 * 224;
#pragma unroll
      for (int k0 = 0; k0 < 224; k0 += 32) {
        v16h a = frag_a_ld(h1s, 224, k0, lane);
        v16h b = frag_b_ld(Bp, 224, k0, lane);
        acc = WMMA_F16(a, b, acc);
      }
      int col = nt * 16 + (lane & 15);
#pragma unroll
      for (int i = 0; i < 8; ++i) {
        int m = i + ((lane < 16) ? 0 : 8);
        zbuf[m * 800 + col] = acc[i] + (float)zp[m * 800 + col];
      }
    }
    __syncthreads();
    // ---- layer 0 gates (i,f,g,o) ----
    for (int idx = tid; idx < 3200; idx += 512) {
      int bb = idx / 200, q = idx - bb * 200;
      const float* zr = &zbuf[bb * 800];
      float zi = zr[q], zf = zr[200 + q], zg = zr[400 + q], zo = zr[600 + q];
      float c = sigmf_(zf) * c1s[idx] + sigmf_(zi) * tanhf(zg);
      float h = sigmf_(zo) * tanhf(c);
      c1s[idx] = c;
      h1s[bb * 224 + q] = (half_t)h;
    }
    __syncthreads();
    // ---- layer 1: zbuf = h1 @ w_ih1^T + h2 @ w_hh1^T + bias ----
    for (int nt = wv; nt < 50; nt += 16) {
      v8f acc = {};
      const half_t* Bp1 = wih1f + (long)nt * 16 * 224;
      const half_t* Bp2 = whh1f + (long)nt * 16 * 224;
#pragma unroll
      for (int k0 = 0; k0 < 224; k0 += 32) {
        v16h a = frag_a_ld(h1s, 224, k0, lane);
        v16h b = frag_b_ld(Bp1, 224, k0, lane);
        acc = WMMA_F16(a, b, acc);
      }
#pragma unroll
      for (int k0 = 0; k0 < 224; k0 += 32) {
        v16h a = frag_a_ld(h2s, 224, k0, lane);
        v16h b = frag_b_ld(Bp2, 224, k0, lane);
        acc = WMMA_F16(a, b, acc);
      }
      int col = nt * 16 + (lane & 15);
      float bias = b_ih1[col] + b_hh1[col];
#pragma unroll
      for (int i = 0; i < 8; ++i) {
        int m = i + ((lane < 16) ? 0 : 8);
        zbuf[m * 800 + col] = acc[i] + bias;
      }
    }
    __syncthreads();
    // ---- layer 1 gates + emit h2 ----
    for (int idx = tid; idx < 3200; idx += 512) {
      int bb = idx / 200, q = idx - bb * 200;
      const float* zr = &zbuf[bb * 800];
      float zi = zr[q], zf = zr[200 + q], zg = zr[400 + q], zo = zr[600 + q];
      float c = sigmf_(zf) * c2s[idx] + sigmf_(zi) * tanhf(zg);
      float h = sigmf_(zo) * tanhf(c);
      c2s[idx] = c;
      h2s[bb * 224 + q] = (half_t)h;
      h2seq[((long)t * 16 + bb) * 224 + q] = (half_t)h;
    }
    __syncthreads();
  }
}

// ---------------- stage 6: output MLP (relu(h2@W1^T+b1)@W2^T+b2), transpose store --
// grid 6000 (one t per wave), block 32
__global__ void k_mlp(const half_t* __restrict__ h2seq, const half_t* __restrict__ o1f,
                      const half_t* __restrict__ o2f, const float* __restrict__ out1_b,
                      const float* __restrict__ out2_b, float* __restrict__ out) {
  __shared__ half_t hid[16 * 128];
  const int lane = threadIdx.x;
  const int t = blockIdx.x;
  for (int i = lane; i < 16 * 128; i += 32) hid[i] = (half_t)0.0f;
  __syncthreads();
  const half_t* A = h2seq + (long)t * 16 * 224;
  for (int nt = 0; nt < 7; ++nt) {
    v8f acc = {};
    const half_t* Bp = o1f + (long)nt * 16 * 224;
#pragma unroll
    for (int k0 = 0; k0 < 224; k0 += 32) {
      v16h a = frag_a_ld(A, 224, k0, lane);
      v16h b = frag_b_ld(Bp, 224, k0, lane);
      acc = WMMA_F16(a, b, acc);
    }
    int col = nt * 16 + (lane & 15);
    float bias = (col < 100) ? out1_b[col] : 0.0f;
#pragma unroll
    for (int i = 0; i < 8; ++i) {
      int m = i + ((lane < 16) ? 0 : 8);
      hid[m * 128 + col] = (half_t)fmaxf(acc[i] + bias, 0.0f);
    }
  }
  __syncthreads();
  for (int nt = 0; nt < 13; ++nt) {
    v8f acc = {};
    const half_t* Bp = o2f + (long)nt * 16 * 128;
#pragma unroll
    for (int k0 = 0; k0 < 128; k0 += 32) {
      v16h a = frag_a_ld(hid, 128, k0, lane);
      v16h b = frag_b_ld(Bp, 128, k0, lane);
      acc = WMMA_F16(a, b, acc);
    }
    int f = nt * 16 + (lane & 15);
    if (f < 200) {
      float bias = out2_b[f];
#pragma unroll
      for (int i = 0; i < 8; ++i) {
        int m = i + ((lane < 16) ? 0 : 8);
        out[((long)m * 200 + f) * 6000 + t] = acc[i] + bias;  // [b][p][t]
      }
    }
  }
}

// ---------------- host launcher ----------------
extern "C" void kernel_launch(void* const* d_in, const int* in_sizes, int n_in,
                              void* d_out, int out_size, void* d_ws, size_t ws_size,
                              hipStream_t stream) {
  (void)in_sizes; (void)n_in; (void)out_size; (void)ws_size;
  const float* x_fmri = (const float*)d_in[0];
  const float* x_meg  = (const float*)d_in[1];
  const float* conv_w = (const float*)d_in[2];
  const float* conv_b = (const float*)d_in[3];
  const float* bn_g   = (const float*)d_in[4];
  const float* bn_b   = (const float*)d_in[5];
  const float* w_ih0  = (const float*)d_in[6];
  const float* w_hh0  = (const float*)d_in[7];
  const float* b_ih0  = (const float*)d_in[8];
  const float* b_hh0  = (const float*)d_in[9];
  const float* w_ih1  = (const float*)d_in[10];
  const float* w_hh1  = (const float*)d_in[11];
  const float* b_ih1  = (const float*)d_in[12];
  const float* b_hh1  = (const float*)d_in[13];
  const float* out1_w = (const float*)d_in[14];
  const float* out1_b = (const float*)d_in[15];
  const float* out2_w = (const float*)d_in[16];
  const float* out2_b = (const float*)d_in[17];
  const float* h0     = (const float*)d_in[18];
  const float* c0     = (const float*)d_in[19];
  float* out = (float*)d_out;

  char* w = (char*)d_ws;
  float*  y     = (float*) (w + 0);           // 16*200*6000 f32   = 76,800,000 B
  half_t* inp16 = (half_t*)(w + 76800000);    // 6000*16*224 f16   = 43,008,000 B
  half_t* h2seq = (half_t*)(w + 119808000);   // 6000*16*224 f16   = 43,008,000 B
  half_t* zpre  = (half_t*)(w + 162816000);   // 96000*800 f16     = 153,600,000 B
  half_t* xT    = (half_t*)(w + 316416000);   // 1008*16*256 f16   = 8,257,536 B
  half_t* wcf   = (half_t*)(w + 324673536);   // 6*5*208*256 f16   = 3,194,880 B
  half_t* wih0f = (half_t*)(w + 327868416);   // 800*224 f16
  half_t* whh0f = (half_t*)(w + 328226816);
  half_t* wih1f = (half_t*)(w + 328585216);
  half_t* whh1f = (half_t*)(w + 328943616);
  half_t* o1f   = (half_t*)(w + 329302016);   // 112*224 f16
  half_t* o2f   = (half_t*)(w + 329352192);   // 208*128 f16
  float*  mu    = (float*) (w + 329405440);   // 200 f32
  float*  rstd  = (float*) (w + 329406240);   // 200 f32

  auto cdiv = [](long a, long b) { return (unsigned)((a + b - 1) / b); };

  k_zero_h<<<cdiv(21504000, 256), 256, 0, stream>>>(inp16, 21504000L);
  k_zero_h<<<cdiv(21504000, 256), 256, 0, stream>>>(h2seq, 21504000L);
  k_pack_xT<<<cdiv(1008L * 16 * 256, 256), 256, 0, stream>>>(x_fmri, xT);
  k_pack_convw<<<cdiv(1597440, 256), 256, 0, stream>>>(conv_w, wcf);
  k_pack_pad<<<cdiv(800 * 224, 256), 256, 0, stream>>>(w_ih0, wih0f, 800, 200, 800, 224);
  k_pack_pad<<<cdiv(800 * 224, 256), 256, 0, stream>>>(w_hh0, whh0f, 800, 200, 800, 224);
  k_pack_pad<<<cdiv(800 * 224, 256), 256, 0, stream>>>(w_ih1, wih1f, 800, 200, 800, 224);
  k_pack_pad<<<cdiv(800 * 224, 256), 256, 0, stream>>>(w_hh1, whh1f, 800, 200, 800, 224);
  k_pack_pad<<<cdiv(112 * 224, 256), 256, 0, stream>>>(out1_w, o1f, 100, 200, 112, 224);
  k_pack_pad<<<cdiv(208 * 128, 256), 256, 0, stream>>>(out2_w, o2f, 200, 100, 208, 128);

  k_convt<<<dim3(13, 1000, 6), 32, 0, stream>>>(xT, wcf, conv_b, y);
  k_bn_stats<<<200, 256, 0, stream>>>(y, mu, rstd);
  k_bn_apply<<<cdiv(19200000, 256), 256, 0, stream>>>(y, mu, rstd, bn_g, bn_b, x_meg, inp16);
  k_zpre<<<dim3(50, 6000), 32, 0, stream>>>(inp16, wih0f, b_ih0, b_hh0, zpre);
  k_lstm<<<1, 512, 0, stream>>>(zpre, whh0f, wih1f, whh1f, b_ih1, b_hh1, h0, c0, h2seq);
  k_mlp<<<6000, 32, 0, stream>>>(h2seq, o1f, o2f, out1_b, out2_b, out);
}
